// MHAtt_5918464934009
// MI455X (gfx1250) — matrix operand order repaired and checked by reading, not compiled, not run on hardware
//
#include <hip/hip_runtime.h>
#include <math.h>

#define HIDDEN 640
#define HEADS  8
#define HEAD   80
#define DPAD   96
#define BATCH  4
#define SEQ    2048

typedef _Float16 h16;
typedef __attribute__((ext_vector_type(16))) _Float16 v16h;
typedef __attribute__((ext_vector_type(8)))  _Float16 v8h;
typedef __attribute__((ext_vector_type(8)))  float    v8f;

static __device__ inline v8f wmma_f16(v16h a, v16h b, v8f c) {
  return __builtin_amdgcn_wmma_f32_16x16x32_f16(false, a, false, b, (short)0, c,
                                                false, false);
}

// A fragment (16x32 f16, M=rows): lane holds row; K = kb + 16*sel + {0..15}
// (sel = lane>>4). f32 source -> convert.
static __device__ inline v16h frag_a_from_f32(const float* row, int kb, int sel) {
  v8f lo = *(const v8f*)(row + kb + 16 * sel);
  v8f hi = *(const v8f*)(row + kb + 16 * sel + 8);
  v16h r;
#pragma unroll
  for (int i = 0; i < 8; ++i) { r[i] = (h16)lo[i]; r[8 + i] = (h16)hi[i]; }
  return r;
}

// B fragment (32x16 f16, N=cols): lane holds col; K halves kb + 16*sel + {0..15}. f32 source.
static __device__ inline v16h frag_b_from_f32(const float* row, int kb, int sel) {
  v8f lo = *(const v8f*)(row + kb + 16 * sel);
  v8f hi = *(const v8f*)(row + kb + 16 * sel + 8);
  v16h r;
#pragma unroll
  for (int i = 0; i < 8; ++i) { r[i] = (h16)lo[i]; r[8 + i] = (h16)hi[i]; }
  return r;
}

// Same fragments from f16 (contraction-contiguous) storage (global or LDS).
static __device__ inline v16h frag_a_from_h(const h16* row, int kb, int sel) {
  union { v16h v; v8h h[2]; } u;
  u.h[0] = *(const v8h*)(row + kb + 16 * sel);
  u.h[1] = *(const v8h*)(row + kb + 16 * sel + 8);
  return u.v;
}
static __device__ inline v16h frag_b_from_h(const h16* row, int kb, int sel) {
  return *(const v16h*)(row + kb + 16 * sel);
}

// Async 16-byte global -> LDS copy (gfx1250, ASYNCcnt-tracked).
// VDST = LDS byte offset (low 32 bits of a generic shared pointer), VADDR = 64-bit global addr.
static __device__ inline void async_copy_b128(const h16* g, h16* l) {
  unsigned int loff = (unsigned int)(unsigned long long)(const void*)l;
  asm volatile("global_load_async_to_lds_b128 %0, %1, off"
               :: "v"(loff), "v"(g)
               : "memory");
}
static __device__ inline void wait_asynccnt0() {
  asm volatile("s_wait_asynccnt 0x0" ::: "memory");
}

// ---------------- zero init (Qh/Kh padding must be 0) ----------------
__global__ void zero_kernel(unsigned int* p, unsigned long long n) {
  unsigned long long i = (unsigned long long)blockIdx.x * blockDim.x + threadIdx.x;
  unsigned long long stride = (unsigned long long)gridDim.x * blockDim.x;
  for (; i < n; i += stride) p[i] = 0u;
}

// ---------------- Q/K projection: D = W x X^T, store [b,h,tok,DPAD] f16 ----------------
__global__ void __launch_bounds__(256) proj_qk_kernel(
    const float* __restrict__ X, const float* __restrict__ W,
    const float* __restrict__ bias, h16* __restrict__ dst) {
  const int lane = threadIdx.x & 31, wave = threadIdx.x >> 5;
  const int sel = lane >> 4, l16 = lane & 15;
  const int m0 = blockIdx.y * 32;               // output-feature base
  const int t0 = blockIdx.x * 256 + wave * 32;  // global token base (0..8191)

  v8f acc[2][2] = {};
  const float* wr0 = W + (size_t)(m0 + l16) * HIDDEN;
  const float* wr1 = W + (size_t)(m0 + 16 + l16) * HIDDEN;
  const float* xr0 = X + (size_t)(t0 + l16) * HIDDEN;
  const float* xr1 = X + (size_t)(t0 + 16 + l16) * HIDDEN;

  for (int kb = 0; kb < HIDDEN; kb += 32) {
    __builtin_prefetch(xr0 + kb + 64, 0, 1);
    __builtin_prefetch(xr1 + kb + 64, 0, 1);
    v16h a0 = frag_a_from_f32(wr0, kb, sel);
    v16h a1 = frag_a_from_f32(wr1, kb, sel);
    v16h b0 = frag_b_from_f32(xr0, kb, sel);
    v16h b1 = frag_b_from_f32(xr1, kb, sel);
    acc[0][0] = wmma_f16(a0, b0, acc[0][0]);
    acc[0][1] = wmma_f16(a0, b1, acc[0][1]);
    acc[1][0] = wmma_f16(a1, b0, acc[1][0]);
    acc[1][1] = wmma_f16(a1, b1, acc[1][1]);
  }

#pragma unroll
  for (int mi = 0; mi < 2; ++mi) {
    const int fbase = m0 + mi * 16 + 8 * sel;  // 8 consecutive feats (never straddles a head)
    const int h = fbase / HEAD;
    const int c = fbase % HEAD;
#pragma unroll
    for (int ni = 0; ni < 2; ++ni) {
      const int tok = t0 + ni * 16 + l16;
      const int b = tok / SEQ, tr = tok % SEQ;
      v8h out;
#pragma unroll
      for (int r = 0; r < 8; ++r) out[r] = (h16)(acc[mi][ni][r] + bias[fbase + r]);
      h16* p = dst + ((size_t)((b * HEADS + h) * SEQ + tr)) * DPAD + c;
      *(v8h*)p = out;
    }
  }
}

// ---------------- V projection: D = X x W^T, store transposed Vt[b,h,d,tok] f16 ----------------
__global__ void __launch_bounds__(256) proj_v_kernel(
    const float* __restrict__ X, const float* __restrict__ W,
    const float* __restrict__ bias, h16* __restrict__ vt) {
  const int lane = threadIdx.x & 31, wave = threadIdx.x >> 5;
  const int sel = lane >> 4, l16 = lane & 15;
  const int n0 = blockIdx.y * 32;               // feature base
  const int t0 = blockIdx.x * 256 + wave * 32;  // token base

  v8f acc[2][2] = {};
  const float* xr0 = X + (size_t)(t0 + l16) * HIDDEN;
  const float* xr1 = X + (size_t)(t0 + 16 + l16) * HIDDEN;
  const float* wr0 = W + (size_t)(n0 + l16) * HIDDEN;
  const float* wr1 = W + (size_t)(n0 + 16 + l16) * HIDDEN;

  for (int kb = 0; kb < HIDDEN; kb += 32) {
    __builtin_prefetch(xr0 + kb + 64, 0, 1);
    __builtin_prefetch(xr1 + kb + 64, 0, 1);
    v16h a0 = frag_a_from_f32(xr0, kb, sel);
    v16h a1 = frag_a_from_f32(xr1, kb, sel);
    v16h b0 = frag_b_from_f32(wr0, kb, sel);
    v16h b1 = frag_b_from_f32(wr1, kb, sel);
    acc[0][0] = wmma_f16(a0, b0, acc[0][0]);
    acc[0][1] = wmma_f16(a0, b1, acc[0][1]);
    acc[1][0] = wmma_f16(a1, b0, acc[1][0]);
    acc[1][1] = wmma_f16(a1, b1, acc[1][1]);
  }

#pragma unroll
  for (int ni = 0; ni < 2; ++ni) {
    const int feat = n0 + ni * 16 + l16;
    const int h = feat / HEAD;
    const int c = feat % HEAD;
    const float bv_ = bias[feat];
#pragma unroll
    for (int ti = 0; ti < 2; ++ti) {
      const int tb = t0 + ti * 16 + 8 * sel;  // 8 consecutive tokens per lane
      const int b = tb / SEQ, tr = tb % SEQ;
      v8h out;
#pragma unroll
      for (int r = 0; r < 8; ++r) out[r] = (h16)(acc[ti][ni][r] + bv_);
      h16* p = vt + ((size_t)((b * HEADS + h) * HEAD + c)) * SEQ + tr;
      *(v8h*)p = out;
    }
  }
}

// ---------------- flash attention: 1 wave = 16 query rows, 64-key blocks ----------------
// K/V tiles staged in LDS with async global->LDS copies, double-buffered.
__global__ void __launch_bounds__(256) attn_kernel(
    const h16* __restrict__ Qh, const h16* __restrict__ Kh,
    const h16* __restrict__ Vt, float* __restrict__ atted) {
  __shared__ __align__(32) h16 ldsK[2][64][DPAD];  // 2 x 12 KB
  __shared__ __align__(32) h16 ldsV[2][HEAD][64];  // 2 x 10 KB
  __shared__ __align__(32) h16 plds[8][16][64];    // 16 KB

  const int tid = threadIdx.x;
  const int lane = tid & 31, wave = tid >> 5;
  const int sel = lane >> 4, l16 = lane & 15;
  const int qb = blockIdx.x & 15;
  const int h = (blockIdx.x >> 4) & 7;
  const int b = blockIdx.x >> 7;
  const int m0 = qb * 128 + wave * 16;  // query row base

  const h16* qbase = Qh + ((size_t)((b * HEADS + h) * SEQ + m0 + l16)) * DPAD;
  const h16* kbase = Kh + ((size_t)((b * HEADS + h) * SEQ)) * DPAD;
  const h16* vbase = Vt + ((size_t)((b * HEADS + h) * HEAD)) * SEQ;

  // cooperative async stage of one 64-key K/V tile
  auto issue_tiles = [&](int buf, int kb) {
    for (int s = tid; s < 64 * 12; s += 256) {         // 64 rows x 192 B
      const int row = s / 12, ch = (s % 12) * 8;
      async_copy_b128(kbase + (size_t)(kb + row) * DPAD + ch, &ldsK[buf][row][ch]);
    }
    for (int s = tid; s < HEAD * 8; s += 256) {        // 80 rows x 128 B
      const int row = s / 8, ch = (s % 8) * 8;
      async_copy_b128(vbase + (size_t)row * SEQ + kb + ch, &ldsV[buf][row][ch]);
    }
  };

  v16h aq[3];
#pragma unroll
  for (int ks = 0; ks < 3; ++ks) aq[ks] = frag_a_from_h(qbase, ks * 32, sel);

  v8f o[5] = {};
  float rm[8], rl[8];
#pragma unroll
  for (int r = 0; r < 8; ++r) { rm[r] = -INFINITY; rl[r] = 0.f; }

  const float scale = 0.1118033988749895f;  // 1/sqrt(80)

  issue_tiles(0, 0);

  for (int kb = 0; kb < SEQ; kb += 64) {
    const int buf = (kb >> 6) & 1;
    wait_asynccnt0();     // my wave's copies into `buf` done
    __syncthreads();      // everyone's copies done; prev iteration's LDS reads retired
    if (kb + 64 < SEQ) issue_tiles(buf ^ 1, kb + 64);

    // S = Q K^T for 4 key tiles (d = 96 -> 3 k-steps each)
    v8f s[4] = {};
#pragma unroll
    for (int j = 0; j < 4; ++j) {
#pragma unroll
      for (int ks = 0; ks < 3; ++ks) {
        v16h bk = frag_b_from_h(&ldsK[buf][j * 16 + l16][0], ks * 32, sel);
        s[j] = wmma_f16(aq[ks], bk, s[j]);
      }
    }

    // online softmax; row r lives across 16 lanes of a half-wave in vgpr r
#pragma unroll
    for (int r = 0; r < 8; ++r) {
      float p[4];
#pragma unroll
      for (int j = 0; j < 4; ++j) p[j] = s[j][r] * scale;
      float t = fmaxf(fmaxf(p[0], p[1]), fmaxf(p[2], p[3]));
#pragma unroll
      for (int mk = 1; mk < 16; mk <<= 1) t = fmaxf(t, __shfl_xor(t, mk, 32));
      const float mnew = fmaxf(rm[r], t);
      const float corr = __expf(rm[r] - mnew);
      rm[r] = mnew;
      rl[r] *= corr;
#pragma unroll
      for (int dt = 0; dt < 5; ++dt) o[dt][r] *= corr;
      float sum = 0.f;
#pragma unroll
      for (int j = 0; j < 4; ++j) { p[j] = __expf(p[j] - mnew); sum += p[j]; s[j][r] = p[j]; }
#pragma unroll
      for (int mk = 1; mk < 16; mk <<= 1) sum += __shfl_xor(sum, mk, 32);
      rl[r] += sum;
    }

    // transpose P (D-layout) -> row-major [16][64] in wave-private LDS
#pragma unroll
    for (int r = 0; r < 8; ++r) {
      const int row = r + 8 * sel;
#pragma unroll
      for (int j = 0; j < 4; ++j) plds[wave][row][j * 16 + l16] = (h16)s[j][r];
    }
    __syncthreads();

    // O += P V  (K-dim = 64 keys -> 2 A fragments)
    const v16h ap0 = frag_a_from_h(&plds[wave][l16][0], 0, sel);
    const v16h ap1 = frag_a_from_h(&plds[wave][l16][0], 32, sel);
#pragma unroll
    for (int dt = 0; dt < 5; ++dt) {
      v16h bv0 = frag_b_from_h(&ldsV[buf][dt * 16 + l16][0], 0, sel);
      v16h bv1 = frag_b_from_h(&ldsV[buf][dt * 16 + l16][0], 32, sel);
      o[dt] = wmma_f16(ap0, bv0, o[dt]);
      o[dt] = wmma_f16(ap1, bv1, o[dt]);
    }
  }

  const int col0 = h * HEAD;
#pragma unroll
  for (int r = 0; r < 8; ++r) {
    const float inv = 1.0f / rl[r];
    const int tok = m0 + r + 8 * sel;
    float* orow = atted + ((size_t)b * SEQ + tok) * HIDDEN + col0;
#pragma unroll
    for (int dt = 0; dt < 5; ++dt) orow[dt * 16 + l16] = o[dt][r] * inv;
  }
}

// ---------------- output projection: D = Wm x atted^T + bm, f32 out ----------------
__global__ void __launch_bounds__(256) out_proj_kernel(
    const float* __restrict__ A, const float* __restrict__ W,
    const float* __restrict__ bias, float* __restrict__ out) {
  const int lane = threadIdx.x & 31, wave = threadIdx.x >> 5;
  const int sel = lane >> 4, l16 = lane & 15;
  const int m0 = blockIdx.y * 32;
  const int t0 = blockIdx.x * 256 + wave * 32;

  v8f acc[2][2] = {};
  const float* wr0 = W + (size_t)(m0 + l16) * HIDDEN;
  const float* wr1 = W + (size_t)(m0 + 16 + l16) * HIDDEN;
  const float* xr0 = A + (size_t)(t0 + l16) * HIDDEN;
  const float* xr1 = A + (size_t)(t0 + 16 + l16) * HIDDEN;

  for (int kb = 0; kb < HIDDEN; kb += 32) {
    __builtin_prefetch(xr0 + kb + 64, 0, 1);
    __builtin_prefetch(xr1 + kb + 64, 0, 1);
    v16h a0 = frag_a_from_f32(wr0, kb, sel);
    v16h a1 = frag_a_from_f32(wr1, kb, sel);
    v16h b0 = frag_b_from_f32(xr0, kb, sel);
    v16h b1 = frag_b_from_f32(xr1, kb, sel);
    acc[0][0] = wmma_f16(a0, b0, acc[0][0]);
    acc[0][1] = wmma_f16(a0, b1, acc[0][1]);
    acc[1][0] = wmma_f16(a1, b0, acc[1][0]);
    acc[1][1] = wmma_f16(a1, b1, acc[1][1]);
  }

#pragma unroll
  for (int mi = 0; mi < 2; ++mi) {
    const int fbase = m0 + mi * 16 + 8 * sel;
#pragma unroll
    for (int ni = 0; ni < 2; ++ni) {
      const int tok = t0 + ni * 16 + l16;
      v8f vals;
#pragma unroll
      for (int r = 0; r < 8; ++r) vals[r] = acc[mi][ni][r] + bias[fbase + r];
      *(v8f*)(out + (size_t)tok * HIDDEN + fbase) = vals;
    }
  }
}

extern "C" void kernel_launch(void* const* d_in, const int* in_sizes, int n_in,
                              void* d_out, int out_size, void* d_ws, size_t ws_size,
                              hipStream_t stream) {
  const float* v  = (const float*)d_in[0];
  const float* k  = (const float*)d_in[1];
  const float* q  = (const float*)d_in[2];
  const float* Wv = (const float*)d_in[3];
  const float* bv = (const float*)d_in[4];
  const float* Wk = (const float*)d_in[5];
  const float* bk = (const float*)d_in[6];
  const float* Wq = (const float*)d_in[7];
  const float* bq = (const float*)d_in[8];
  const float* Wm = (const float*)d_in[9];
  const float* bm = (const float*)d_in[10];
  float* out = (float*)d_out;

  // workspace layout
  const size_t QH_BYTES = (size_t)BATCH * HEADS * SEQ * DPAD * sizeof(h16);  // 12.58 MB
  const size_t VT_BYTES = (size_t)BATCH * HEADS * HEAD * SEQ * sizeof(h16);  // 10.49 MB
  h16*   Qh = (h16*)((char*)d_ws);
  h16*   Kh = (h16*)((char*)d_ws + QH_BYTES);
  h16*   Vt = (h16*)((char*)d_ws + 2 * QH_BYTES);
  float* At = (float*)((char*)d_ws + 2 * QH_BYTES + VT_BYTES);

  // 1) zero Qh/Kh (padding columns must be 0)
  const unsigned long long zwords = (unsigned long long)(2 * QH_BYTES / 4);
  zero_kernel<<<2048, 256, 0, stream>>>((unsigned int*)d_ws, zwords);

  // 2) projections
  dim3 gproj(32, 20);
  proj_qk_kernel<<<gproj, 256, 0, stream>>>(q, Wq, bq, Qh);
  proj_qk_kernel<<<gproj, 256, 0, stream>>>(k, Wk, bk, Kh);
  proj_v_kernel<<<gproj, 256, 0, stream>>>(v, Wv, bv, Vt);

  // 3) flash attention: 512 workgroups (b*h*16 query blocks of 128)
  attn_kernel<<<512, 256, 0, stream>>>(Qh, Kh, Vt, At);

  // 4) output projection
  out_proj_kernel<<<gproj, 256, 0, stream>>>(At, Wm, bm, out);
}
